// LogSigRNNEncoder_65025804862074
// MI455X (gfx1250) — compile-verified, matrix operand order checked by reading, not run on hardware
//
#include <hip/hip_runtime.h>
#include <hip/hip_bf16.h>
#include <stdint.h>

// LogSig-RNN encoder, restructured for MI455X (gfx1250):
//  - update mask fires only at t in {0,50,100,150,199}  => scan collapses to
//    5 sequential GEMM stages of M = 256*seglen rows through the FFN chain.
//  - ~109 GFLOP dense GEMM -> v_wmma_f32_16x16x32_f16 (f16 in, f32 accum).
//  - GEMM tiles staged into LDS by the Tensor Data Mover (tensor_load_to_lds),
//    double buffered, synced via s_wait_tensorcnt + workgroup barrier.

typedef _Float16 f16;
typedef __attribute__((ext_vector_type(16))) _Float16 v16h;
typedef __attribute__((ext_vector_type(8)))  _Float16 v8h;
typedef __attribute__((ext_vector_type(8)))  float    v8f;
typedef __attribute__((ext_vector_type(4)))  uint32_t v4u;
typedef __attribute__((ext_vector_type(8)))  int      v8i;
typedef __attribute__((ext_vector_type(4)))  int      v4i;

#define B_SZ 256
#define W_SZ 200
#define DIN  8
#define HID  512
#define CHP  64     // logsig feature width padded (36 -> 64)
#define FIN  576    // ffn input padded (548 -> 576)
#define TPAD 40     // LDS tile row stride in halves (32 data + 8 pad)

// ---------------------------------------------------------------- TDM helper
__device__ __forceinline__ uint32_t lds_off(const void* p) {
  // flat LDS address: aperture in [63:32], byte offset in [31:0]
  return (uint32_t)(uintptr_t)p;
}

// Load a 128-row x 32-half tile (row stride = row_stride_elems halves) from
// global into LDS at lds_byte_off, padding 4 DWORDs after every 16 DWORDs so
// the LDS row stride becomes TPAD(=40) halves.
__device__ __forceinline__ void tdm_load_tile(uint32_t lds_byte_off,
                                              const f16* gptr,
                                              uint32_t row_stride_elems) {
  const uint64_t ga = (uint64_t)(uintptr_t)gptr;
  v4u g0;
  g0[0] = 1u;                                            // count=1, user mode
  g0[1] = lds_byte_off;                                  // lds_addr
  g0[2] = (uint32_t)ga;                                  // global_addr[31:0]
  g0[3] = (uint32_t)((ga >> 32) & 0x01FFFFFFu) | (2u << 30); // addr hi | type=2
  const uint32_t TD0 = 1u << 30, TD1 = 1u << 30;         // huge tensor dims (no OOB)
  v8i g1;
  g1[0] = (int)((1u << 16)        // data_size = 1 -> 2 bytes
              | (1u << 20)        // pad_enable
              | (3u << 22)        // pad_interval: every 16 DWORDs
              | (3u << 25));      // pad_amount: 4 DWORDs
  g1[1] = (int)((TD0 & 0xFFFFu) << 16);                          // tensor_dim0 lo
  g1[2] = (int)(((TD0 >> 16) & 0xFFFFu) | ((TD1 & 0xFFFFu) << 16));
  g1[3] = (int)(((TD1 >> 16) & 0xFFFFu) | (32u << 16));          // tile_dim0 = 32
  g1[4] = (int)128;                                              // tile_dim1 = 128
  g1[5] = (int)row_stride_elems;                                 // dim0_stride lo
  g1[6] = 0;
  g1[7] = 0;
  v4i z4 = {0, 0, 0, 0};
  v8i z8 = {0, 0, 0, 0, 0, 0, 0, 0};
  __builtin_amdgcn_tensor_load_to_lds(g0, g1, z4, z4, z8, 0);
}

// ------------------------------------------------------------ fragment load
// 16-bit A/B fragment: lane<16 holds K 0..7 & 16..23, lane>=16 holds 8..15 & 24..31
__device__ __forceinline__ v16h frag_ld(const f16* rowp, int khalf) {
  const v8h lo = *(const v8h*)(rowp + khalf);
  const v8h hi = *(const v8h*)(rowp + khalf + 16);
  v16h r;
#pragma unroll
  for (int q = 0; q < 8; ++q) { r[q] = lo[q]; r[8 + q] = hi[q]; }
  return r;
}

// ------------------------------------------------------------------- logsig
__global__ __launch_bounds__(64) void k_logsig(const float* __restrict__ P,
                                               f16* __restrict__ ls) {
  __shared__ float p[W_SZ * DIN];
  const int b = blockIdx.x;
  const int tid = threadIdx.x;
  for (int i = tid; i < W_SZ * DIN; i += 64)
    p[i] = P[(size_t)b * W_SZ * DIN + i];
  __syncthreads();
  // dense increment features + zero padding (+ full zero row at t=0)
  for (int i = tid; i < W_SZ * CHP; i += 64) {
    const int t = i >> 6, j = i & 63;
    if (j < DIN || j >= 36 || t == 0) {
      const float v = (j < DIN && t > 0) ? p[t * DIN + j] : 0.0f;
      ls[((size_t)b * W_SZ + t) * CHP + j] = (f16)v;
    }
  }
  // level-2 terms: running antisymmetric cumsum with checkpoints at 50/100/150
  if (tid < 28) {
    int ii = 0, jj = 1, c = tid;
    for (int i = 0; i < DIN - 1; ++i) {
      const int cnt = DIN - 1 - i;
      if (c < cnt) { ii = i; jj = i + 1 + c; break; }
      c -= cnt;
    }
    float run = 0.0f, ref = 0.0f;
    for (int s = 0; s < W_SZ - 1; ++s) {
      const float pi = p[s * DIN + ii],       pj = p[s * DIN + jj];
      const float qi = p[(s + 1) * DIN + ii], qj = p[(s + 1) * DIN + jj];
      run += 0.5f * (pi * (qj - pj) - (qi - pi) * pj);
      const int t = s + 1;
      if (t == 50 || t == 100 || t == 150) ref = run;
      ls[((size_t)b * W_SZ + t) * CHP + DIN + tid] = (f16)(run - ref);
    }
  }
}

// -------------------------------------------------------- weight conversion
__global__ void k_cvt(const float* __restrict__ src, f16* __restrict__ dst,
                      int N, int K, int Kp) {
  const int i = blockIdx.x * blockDim.x + threadIdx.x;
  if (i >= N * Kp) return;
  const int n = i / Kp, k = i - n * Kp;
  dst[i] = (f16)(k < K ? src[(size_t)n * K + k] : 0.0f);
}

// ------------------------------------------------------------------ packing
__global__ __launch_bounds__(256) void k_pack(const f16* __restrict__ state,
                                              const f16* __restrict__ ls,
                                              f16* __restrict__ X,
                                              int t0, int segLen) {
  const int r = blockIdx.x;
  const int b = r / segLen;
  const int t = t0 + (r - b * segLen);
  f16* xr = X + (size_t)r * FIN;
  const f16* st = state + (size_t)b * HID;
  const f16* lr = ls + ((size_t)b * W_SZ + t) * CHP;
  for (int k = threadIdx.x; k < FIN; k += 256)
    xr[k] = (k < HID) ? st[k] : lr[k - HID];
}

// --------------------------------------------------------------- WMMA GEMM
// out[m][n] = act( sum_k A[m][k] * Wt[n][k] + bias[n] )
// act: 0 = relu -> f16 out16, 1 = tanh -> f16 out16, 2 = none -> f32 out32
//      (out32 indexed as final output [b][t0 + m%segLen][n])
__global__ __launch_bounds__(256) void k_gemm(const f16* __restrict__ A,
                                              const f16* __restrict__ Wt,
                                              const float* __restrict__ bias,
                                              f16* __restrict__ out16,
                                              float* __restrict__ out32,
                                              int M, int K, int act,
                                              int t0, int segLen) {
  __shared__ __align__(16) f16 Xs[2][128 * TPAD];
  __shared__ __align__(16) f16 Ws[2][128 * TPAD];
  const int nk = K >> 5;
  const int rowBase = blockIdx.y * 128;
  const int colBase = blockIdx.x * 128;
  const int wid  = threadIdx.x >> 5;
  const int lane = threadIdx.x & 31;
  const int wr = (wid & 3) * 32;    // wave's M slice within block tile
  const int wc = (wid >> 2) * 64;   // wave's N slice within block tile
  const int lrow  = lane & 15;
  const int khalf = (lane >> 4) * 8;

  if (wid == 0) {
    tdm_load_tile(lds_off(&Xs[0][0]), A  + (size_t)rowBase * K, (uint32_t)K);
    tdm_load_tile(lds_off(&Ws[0][0]), Wt + (size_t)colBase * K, (uint32_t)K);
  }

  const v8f vzero = {0.f, 0.f, 0.f, 0.f, 0.f, 0.f, 0.f, 0.f};
  v8f acc[2][4];
#pragma unroll
  for (int i = 0; i < 2; ++i)
#pragma unroll
    for (int j = 0; j < 4; ++j) acc[i][j] = vzero;

  for (int kb = 0; kb < nk; ++kb) {
    const int cur = kb & 1;
    if (wid == 0) {
      if (kb + 1 < nk) {
        const int nxt = cur ^ 1;
        tdm_load_tile(lds_off(&Xs[nxt][0]),
                      A + (size_t)rowBase * K + (size_t)(kb + 1) * 32,
                      (uint32_t)K);
        tdm_load_tile(lds_off(&Ws[nxt][0]),
                      Wt + (size_t)colBase * K + (size_t)(kb + 1) * 32,
                      (uint32_t)K);
        __builtin_amdgcn_s_wait_tensorcnt(2);  // oldest 2 (current tiles) done
      } else {
        __builtin_amdgcn_s_wait_tensorcnt(0);
      }
    }
    __syncthreads();

    // preload ALL fragments (2 A + 4 B) into distinct regs so the 8 WMMAs
    // issue back-to-back after a single dscnt wait (no WAR reload bubble)
    v16h afr[2];
#pragma unroll
    for (int mi = 0; mi < 2; ++mi)
      afr[mi] = frag_ld(&Xs[cur][(wr + mi * 16 + lrow) * TPAD], khalf);
    v16h bfr[4];
#pragma unroll
    for (int ni = 0; ni < 4; ++ni)
      bfr[ni] = frag_ld(&Ws[cur][(wc + ni * 16 + lrow) * TPAD], khalf);
#pragma unroll
    for (int mi = 0; mi < 2; ++mi)
#pragma unroll
      for (int ni = 0; ni < 4; ++ni)
        acc[mi][ni] = __builtin_amdgcn_wmma_f32_16x16x32_f16(
            false, afr[mi], false, bfr[ni], (short)0, acc[mi][ni], false, false);
    __syncthreads();
  }

  // C/D layout: lane(0..15) -> N=lane, VGPR q -> M=q ; lane(16..31) -> M=q+8
  const int nloc = lane & 15;
  const int mhi  = (lane >> 4) * 8;
#pragma unroll
  for (int mi = 0; mi < 2; ++mi) {
#pragma unroll
    for (int ni = 0; ni < 4; ++ni) {
      const int col = colBase + wc + ni * 16 + nloc;
      const float bb = bias ? bias[col] : 0.0f;
#pragma unroll
      for (int q = 0; q < 8; ++q) {
        const int m = rowBase + wr + mi * 16 + mhi + q;
        float v = acc[mi][ni][q] + bb;
        if (act == 0)      v = v > 0.0f ? v : 0.0f;
        else if (act == 1) v = tanhf(v);
        if (out16) out16[(size_t)m * HID + col] = (f16)v;
        if (out32) {
          const int b = m / segLen;
          const int t = t0 + (m - b * segLen);
          out32[((size_t)b * W_SZ + t) * HID + col] = v;
        }
      }
    }
  }
}

// --------------------------------------------------------------- state copy
__global__ __launch_bounds__(512) void k_update(f16* __restrict__ state,
                                                const f16* __restrict__ H,
                                                int segLen) {
  const int b = blockIdx.x;
  state[(size_t)b * HID + threadIdx.x] =
      H[(size_t)(b * segLen + segLen - 1) * HID + threadIdx.x];
}

// -------------------------------------------------------------------- host
extern "C" void kernel_launch(void* const* d_in, const int* in_sizes, int n_in,
                              void* d_out, int out_size, void* d_ws, size_t ws_size,
                              hipStream_t stream) {
  (void)in_sizes; (void)n_in; (void)out_size; (void)ws_size;
  const float* real = (const float*)d_in[0];
  const float* W1 = (const float*)d_in[1];
  const float* b1 = (const float*)d_in[2];
  const float* W2 = (const float*)d_in[3];
  const float* b2 = (const float*)d_in[4];
  const float* W3 = (const float*)d_in[5];
  const float* b3 = (const float*)d_in[6];
  const float* Wl = (const float*)d_in[7];
  float* out = (float*)d_out;

  char* ws = (char*)d_ws;
  size_t off = 0;
  auto alloc = [&](size_t bytes) -> void* {
    void* p = ws + off;
    off = (off + bytes + 255) & ~(size_t)255;
    return p;
  };
  f16* ls    = (f16*)alloc((size_t)B_SZ * W_SZ * CHP * 2);
  f16* W1h   = (f16*)alloc((size_t)HID * FIN * 2);
  f16* W2h   = (f16*)alloc((size_t)HID * HID * 2);
  f16* W3h   = (f16*)alloc((size_t)HID * HID * 2);
  f16* Wlh   = (f16*)alloc((size_t)HID * HID * 2);
  f16* state = (f16*)alloc((size_t)B_SZ * HID * 2);
  f16* X     = (f16*)alloc((size_t)B_SZ * 50 * FIN * 2);
  f16* Z1    = (f16*)alloc((size_t)B_SZ * 50 * HID * 2);
  f16* Z2    = (f16*)alloc((size_t)B_SZ * 50 * HID * 2);
  f16* H     = X;  // X dead after layer-1 GEMM; reuse for tanh output

  (void)hipMemsetAsync(state, 0, (size_t)B_SZ * HID * 2, stream);
  k_logsig<<<B_SZ, 64, 0, stream>>>(real, ls);
  {
    int n = HID * FIN;
    k_cvt<<<(n + 255) / 256, 256, 0, stream>>>(W1, W1h, HID, HID + 36, FIN);
    n = HID * HID;
    k_cvt<<<(n + 255) / 256, 256, 0, stream>>>(W2, W2h, HID, HID, HID);
    k_cvt<<<(n + 255) / 256, 256, 0, stream>>>(W3, W3h, HID, HID, HID);
    k_cvt<<<(n + 255) / 256, 256, 0, stream>>>(Wl, Wlh, HID, HID, HID);
  }

  // 5 sequential stages: state is piecewise constant between mask hits.
  const int t0s[5]  = {0, 1, 51, 101, 151};
  const int lens[5] = {1, 50, 50, 50, 49};
  for (int s = 0; s < 5; ++s) {
    const int t0 = t0s[s], len = lens[s], M = B_SZ * len;
    k_pack<<<M, 256, 0, stream>>>(state, ls, X, t0, len);
    dim3 g(HID / 128, M / 128);
    k_gemm<<<g, 256, 0, stream>>>(X,  W1h, b1,      Z1,      nullptr, M, FIN, 0, 0, 1);
    k_gemm<<<g, 256, 0, stream>>>(Z1, W2h, b2,      Z2,      nullptr, M, HID, 0, 0, 1);
    k_gemm<<<g, 256, 0, stream>>>(Z2, W3h, b3,      H,       nullptr, M, HID, 1, 0, 1);
    k_gemm<<<g, 256, 0, stream>>>(H,  Wlh, nullptr, nullptr, out,     M, HID, 2, t0, len);
    if (s < 4) k_update<<<B_SZ, HID, 0, stream>>>(state, H, len);
  }
}
